// MiMoV2Moe_30133490548817
// MI455X (gfx1250) — matrix operand order repaired and checked by reading, not compiled, not run on hardware
//
#include <hip/hip_runtime.h>

#define TOK 2048
#define HID 1024
#define NEXP 64
#define FFN 256
#define TOPK 8
#define CAP 1024

// ---- optional CDNA5 async global->LDS path (compile-safe feature probe) ----
#if defined(__has_builtin)
#  if __has_builtin(__builtin_amdgcn_global_load_async_to_lds_b128) && \
      __has_builtin(__builtin_amdgcn_s_wait_asynccnt)
#    define USE_ASYNC_X 1
#  endif
#endif
#ifndef USE_ASYNC_X
#  define USE_ASYNC_X 0
#endif

typedef __attribute__((ext_vector_type(4))) int i32x4;
#if USE_ASYNC_X
typedef __attribute__((address_space(1))) i32x4* gv4i;   // global int4*
typedef __attribute__((address_space(3))) i32x4* lv4i;   // LDS int4*
#endif

typedef __attribute__((ext_vector_type(16))) __bf16 bf16x16;
typedef __attribute__((ext_vector_type(8)))  __bf16 bf16x8;
typedef __attribute__((ext_vector_type(4)))  __bf16 bf16x4;
typedef __attribute__((ext_vector_type(8)))  float  f32x8;

__device__ __forceinline__ bf16x16 frag_cat(bf16x8 lo, bf16x8 hi) {
  return __builtin_shufflevector(lo, hi, 0,1,2,3,4,5,6,7,8,9,10,11,12,13,14,15);
}
// A fragment (16-bit A 16x32 layout): lane<16 holds K {0..7, 16..23}, lane>=16 holds K {8..15, 24..31}
__device__ __forceinline__ bf16x16 load_fragA(const __bf16* p) {
  return frag_cat(*(const bf16x8*)p, *(const bf16x8*)(p + 16));
}
// B fragment (same K order as A): lane<16 holds K {0..7, 16..23}, lane>=16 holds K {8..15, 24..31}
__device__ __forceinline__ bf16x16 load_fragB(const __bf16* p) {
  return frag_cat(*(const bf16x8*)p, *(const bf16x8*)(p + 16));
}
__device__ __forceinline__ f32x8 wmma_bf16(bf16x16 a, bf16x16 b, f32x8 c) {
  return __builtin_amdgcn_wmma_f32_16x16x32_bf16(false, a, false, b, (short)0, c, false, false);
}
// transpose-store a 4-row x 4-col fp32 block as bf16 into [col][32 rows] LDS layout
__device__ __forceinline__ void store_t4(__bf16* B, int c, int r0,
                                         float4 a0, float4 a1, float4 a2, float4 a3) {
  bf16x4 t;
  t[0]=(__bf16)a0.x; t[1]=(__bf16)a1.x; t[2]=(__bf16)a2.x; t[3]=(__bf16)a3.x;
  *(bf16x4*)(B + (c + 0) * 32 + r0) = t;
  t[0]=(__bf16)a0.y; t[1]=(__bf16)a1.y; t[2]=(__bf16)a2.y; t[3]=(__bf16)a3.y;
  *(bf16x4*)(B + (c + 1) * 32 + r0) = t;
  t[0]=(__bf16)a0.z; t[1]=(__bf16)a1.z; t[2]=(__bf16)a2.z; t[3]=(__bf16)a3.z;
  *(bf16x4*)(B + (c + 2) * 32 + r0) = t;
  t[0]=(__bf16)a0.w; t[1]=(__bf16)a1.w; t[2]=(__bf16)a2.w; t[3]=(__bf16)a3.w;
  *(bf16x4*)(B + (c + 3) * 32 + r0) = t;
}

// ---------------------------------------------------------------- zero
__global__ void moe_zero_kernel(float* __restrict__ y, int* __restrict__ counts) {
  int idx = blockIdx.x * blockDim.x + threadIdx.x;
  ((float4*)y)[idx] = make_float4(0.f, 0.f, 0.f, 0.f);   // TOK*HID/4 exactly
  if (blockIdx.x == 0 && threadIdx.x < NEXP) counts[threadIdx.x] = 0;
}

// ---------------------------------------------------------------- router
__global__ void moe_router_kernel(const float* __restrict__ x,
                                  const float* __restrict__ gw,
                                  int* __restrict__ ids_out,
                                  int* __restrict__ counts,
                                  int* __restrict__ tlist,
                                  float* __restrict__ wlist) {
  int tid = threadIdx.x;
  int lane = tid & 31;
  int wave = tid >> 5;
  int token = blockIdx.x * 8 + wave;
  const float* xrow = x + (size_t)token * HID;

  float l0 = 0.f, l1 = 0.f;
  #pragma unroll 8
  for (int h = 0; h < HID; ++h) {
    float xv = xrow[h];
    l0 = fmaf(xv, gw[h * NEXP + lane], l0);
    l1 = fmaf(xv, gw[h * NEXP + 32 + lane], l1);
  }
  // softmax over 64 logits spread 2/lane
  float m = fmaxf(l0, l1);
  for (int o = 16; o >= 1; o >>= 1) m = fmaxf(m, __shfl_xor(m, o, 32));
  float p0 = __expf(l0 - m), p1 = __expf(l1 - m);
  float s = p0 + p1;
  for (int o = 16; o >= 1; o >>= 1) s += __shfl_xor(s, o, 32);
  float rs = __builtin_amdgcn_rcpf(s);
  p0 *= rs; p1 *= rs;

  // greedy top-8 (descending, matches lax.top_k)
  float q0 = p0, q1 = p1;
  float wk[TOPK]; int ik[TOPK]; float wsum = 0.f;
  #pragma unroll
  for (int k = 0; k < TOPK; ++k) {
    float bv; int bi;
    if (q0 >= q1) { bv = q0; bi = lane; } else { bv = q1; bi = lane + 32; }
    for (int o = 16; o >= 1; o >>= 1) {
      float ov = __shfl_xor(bv, o, 32);
      int   oi = __shfl_xor(bi, o, 32);
      if (ov > bv || (ov == bv && oi < bi)) { bv = ov; bi = oi; }
    }
    wk[k] = bv; ik[k] = bi; wsum += bv;
    if (bi < 32) { if (lane == bi)        q0 = -1.f; }
    else         { if (lane == (bi - 32)) q1 = -1.f; }
  }
  if (lane == 0) {
    float inv = __builtin_amdgcn_rcpf(wsum);
    #pragma unroll
    for (int k = 0; k < TOPK; ++k) {
      int e = ik[k];
      ids_out[token * TOPK + k] = e;
      int slot = atomicAdd(&counts[e], 1);
      if (slot < CAP) {
        tlist[e * CAP + slot] = token;
        wlist[e * CAP + slot] = wk[k] * inv;
      }
    }
  }
}

// ---------------------------------------------------------------- expert MLP (M = 32 tokens / WG)
// LDS map (bytes): [0,256) tokS/wS | [256,65792) Xb 32x1024 bf16
// stage1: Bgt @65792 (256x32 bf16 T), But @82176 (256x32 bf16 T)      -> end  98560
// stage2: Hb  @65792 (32x256 bf16, aliases Bgt), Bdt @82176 (512x32)  -> end 114944
// async X staging (transient, pre stage-1): Xstg @65792, 8 rows fp32 = 32KB
#define SMEM_BYTES 114944

__global__ void __launch_bounds__(256)
moe_expert_kernel(const float* __restrict__ x,
                  const float* __restrict__ wg,
                  const float* __restrict__ wu,
                  const float* __restrict__ wd,
                  const int* __restrict__ counts,
                  const int* __restrict__ tlist,
                  const float* __restrict__ wlist,
                  float* __restrict__ y) {
  extern __shared__ char smem[];
  int*    tokS = (int*)smem;                    // 32
  float*  wS   = (float*)(smem + 128);          // 32
  __bf16* Xb   = (__bf16*)(smem + 256);
  __bf16* Bgt  = (__bf16*)(smem + 65792);
  __bf16* But  = (__bf16*)(smem + 82176);
  __bf16* Hb   = (__bf16*)(smem + 65792);
  __bf16* Bdt  = (__bf16*)(smem + 82176);

  int e  = blockIdx.x;
  int t0 = blockIdx.y * 32;
  int cnt = counts[e]; if (cnt > CAP) cnt = CAP;
  if (t0 >= cnt) return;

  int tid = threadIdx.x, lane = tid & 31, wave = tid >> 5;
  int nlo  = lane & 15;
  int koff = (lane >> 4) * 8;    // A-frag K sub-offset
  int bko  = (lane >> 4) * 8;    // B-frag K sub-offset
  int rhi  = (lane >> 4) << 3;   // D-frag row offset

  if (tid < 32) {
    int idx = t0 + tid; int tk = 0; float w = 0.f;
    if (idx < cnt) { tk = tlist[e * CAP + idx]; w = wlist[e * CAP + idx]; }
    tokS[tid] = tk; wS[tid] = w;
  }
  __syncthreads();

  // X tile (32 x 1024 fp32) -> bf16 LDS row-major
#if USE_ASYNC_X
  {
    float* Xstg = (float*)(smem + 65792);       // 8 rows fp32, transient
    for (int ch = 0; ch < 4; ++ch) {
      __syncthreads();
      #pragma unroll
      for (int it = 0; it < 8; ++it) {
        int q = tid + it * 256;                 // 2048 float4 per 8-row chunk
        int r = q >> 8;
        int c = (q & 255) << 2;
        const float* src = x + (size_t)tokS[ch * 8 + r] * HID + c;
        float* dst = Xstg + r * HID + c;
        __builtin_amdgcn_global_load_async_to_lds_b128(
            (gv4i)(uintptr_t)src, (lv4i)(unsigned)(uintptr_t)dst, 0, 0);
      }
      __builtin_amdgcn_s_wait_asynccnt(0);
      __syncthreads();
      #pragma unroll
      for (int it = 0; it < 8; ++it) {
        int q = tid + it * 256;
        int r = q >> 8;
        int c = (q & 255) << 2;
        const float4 v = *(const float4*)(Xstg + r * HID + c);
        bf16x4 b; b[0] = (__bf16)v.x; b[1] = (__bf16)v.y; b[2] = (__bf16)v.z; b[3] = (__bf16)v.w;
        *(bf16x4*)(Xb + (ch * 8 + r) * HID + c) = b;
      }
    }
  }
#else
  #pragma unroll
  for (int it = 0; it < 32; ++it) {
    int q = tid + it * 256;           // 8192 float4
    int r = q >> 8;                   // 256 float4 per row
    int c = (q & 255) << 2;
    const float4 v = *(const float4*)(x + (size_t)tokS[r] * HID + c);
    bf16x4 b; b[0] = (__bf16)v.x; b[1] = (__bf16)v.y; b[2] = (__bf16)v.z; b[3] = (__bf16)v.w;
    *(bf16x4*)(Xb + r * HID + c) = b;
  }
#endif

  const float* wg_e = wg + (size_t)e * HID * FFN;
  const float* wu_e = wu + (size_t)e * HID * FFN;
  const float* wd_e = wd + (size_t)e * FFN * HID;

  // ---- stage 1: g = X@Wg, u = X@Wu   (32 x 256, K = 1024)
  f32x8 accg[2][2] = {}, accu[2][2] = {};
  for (int kb = 0; kb < 32; ++kb) {
    __syncthreads();
    if (kb < 31) {
      __builtin_prefetch(wg_e + (size_t)(kb + 1) * 32 * FFN, 0, 0);
      __builtin_prefetch(wu_e + (size_t)(kb + 1) * 32 * FFN, 0, 0);
    }
    // stage B rows kb*32..+32 of [H][FFN], transposed, 4x4 blocks
    #pragma unroll
    for (int it = 0; it < 2; ++it) {
      int gq = tid + it * 256;        // 512 blocks
      int r0 = (gq >> 6) << 2;
      int c  = (gq & 63) << 2;
      const float* pg = wg_e + (size_t)(kb * 32 + r0) * FFN + c;
      const float* pu = wu_e + (size_t)(kb * 32 + r0) * FFN + c;
      float4 g0 = *(const float4*)(pg);
      float4 g1 = *(const float4*)(pg + FFN);
      float4 g2 = *(const float4*)(pg + 2 * FFN);
      float4 g3 = *(const float4*)(pg + 3 * FFN);
      store_t4(Bgt, c, r0, g0, g1, g2, g3);
      float4 u0 = *(const float4*)(pu);
      float4 u1 = *(const float4*)(pu + FFN);
      float4 u2 = *(const float4*)(pu + 2 * FFN);
      float4 u3 = *(const float4*)(pu + 3 * FFN);
      store_t4(But, c, r0, u0, u1, u2, u3);
    }
    __syncthreads();
    bf16x16 a0 = load_fragA(Xb + (size_t)nlo * HID + kb * 32 + koff);
    bf16x16 a1 = load_fragA(Xb + (size_t)(16 + nlo) * HID + kb * 32 + koff);
    #pragma unroll
    for (int tt = 0; tt < 2; ++tt) {
      int col = wave * 32 + tt * 16 + nlo;
      bf16x16 bg = load_fragB(Bgt + col * 32 + bko);
      bf16x16 bu = load_fragB(But + col * 32 + bko);
      accg[0][tt] = wmma_bf16(a0, bg, accg[0][tt]);
      accg[1][tt] = wmma_bf16(a1, bg, accg[1][tt]);
      accu[0][tt] = wmma_bf16(a0, bu, accu[0][tt]);
      accu[1][tt] = wmma_bf16(a1, bu, accu[1][tt]);
    }
  }
  __syncthreads();

  // h = silu(g) * u * combine_w -> bf16 LDS (32 x 256)
  #pragma unroll
  for (int mt = 0; mt < 2; ++mt) {
    #pragma unroll
    for (int tt = 0; tt < 2; ++tt) {
      int n0 = wave * 32 + tt * 16;
      #pragma unroll
      for (int r = 0; r < 8; ++r) {
        int row = mt * 16 + r + rhi;
        float g = accg[mt][tt][r], u = accu[mt][tt][r];
        float sig = __builtin_amdgcn_rcpf(1.f + __expf(-g));
        float hsw = g * sig * u * wS[row];
        Hb[row * FFN + n0 + nlo] = (__bf16)hsw;
      }
    }
  }
  __syncthreads();

  // ---- stage 2: y += h @ Wd   (32 x 1024, K = 256), two 512-col halves
  for (int nh = 0; nh < 2; ++nh) {
    f32x8 acc[2][4] = {};
    for (int kb = 0; kb < 8; ++kb) {
      __syncthreads();
      if (kb < 7) __builtin_prefetch(wd_e + (size_t)(kb + 1) * 32 * HID + nh * 512, 0, 0);
      #pragma unroll
      for (int it = 0; it < 4; ++it) {
        int gq = tid + it * 256;      // 1024 blocks
        int r0 = (gq >> 7) << 2;
        int c  = (gq & 127) << 2;
        const float* pd = wd_e + (size_t)(kb * 32 + r0) * HID + nh * 512 + c;
        float4 d0 = *(const float4*)(pd);
        float4 d1 = *(const float4*)(pd + HID);
        float4 d2 = *(const float4*)(pd + 2 * HID);
        float4 d3 = *(const float4*)(pd + 3 * HID);
        store_t4(Bdt, c, r0, d0, d1, d2, d3);
      }
      __syncthreads();
      bf16x16 a0 = load_fragA(Hb + nlo * FFN + kb * 32 + koff);
      bf16x16 a1 = load_fragA(Hb + (16 + nlo) * FFN + kb * 32 + koff);
      #pragma unroll
      for (int tt = 0; tt < 4; ++tt) {
        int col = wave * 64 + tt * 16 + nlo;
        bf16x16 b = load_fragB(Bdt + col * 32 + bko);
        acc[0][tt] = wmma_bf16(a0, b, acc[0][tt]);
        acc[1][tt] = wmma_bf16(a1, b, acc[1][tt]);
      }
    }
    #pragma unroll
    for (int mt = 0; mt < 2; ++mt) {
      #pragma unroll
      for (int tt = 0; tt < 4; ++tt) {
        int n0 = nh * 512 + wave * 64 + tt * 16 + nlo;
        #pragma unroll
        for (int r = 0; r < 8; ++r) {
          int row = mt * 16 + r + rhi;  // invalid rows carry weight 0 -> add 0
          unsafeAtomicAdd(y + (size_t)tokS[row] * HID + n0, acc[mt][tt][r]);
        }
      }
    }
  }
}

// ---------------------------------------------------------------- launch
extern "C" void kernel_launch(void* const* d_in, const int* in_sizes, int n_in,
                              void* d_out, int out_size, void* d_ws, size_t ws_size,
                              hipStream_t stream) {
  const float* x  = (const float*)d_in[0];
  const float* gw = (const float*)d_in[1];
  const float* wg = (const float*)d_in[2];
  const float* wu = (const float*)d_in[3];
  const float* wd = (const float*)d_in[4];

  float* y   = (float*)d_out;
  int*   ids = (int*)(y + (size_t)TOK * HID);

  int*   counts = (int*)d_ws;
  int*   tlist  = (int*)((char*)d_ws + 1024);
  float* wlist  = (float*)((char*)d_ws + 1024 + (size_t)NEXP * CAP * 4);

  moe_zero_kernel<<<TOK * HID / 4 / 256, 256, 0, stream>>>(y, counts);
  moe_router_kernel<<<TOK / 8, 256, 0, stream>>>(x, gw, ids, counts, tlist, wlist);
  moe_expert_kernel<<<dim3(NEXP, CAP / 32), 256, SMEM_BYTES, stream>>>(
      x, wg, wu, wd, counts, tlist, wlist, y);
}
